// JointSupervisedGroundedCoreferencer_62010737819967
// MI455X (gfx1250) — matrix operand order repaired and checked by reading, not compiled, not run on hardware
//
#include <hip/hip_runtime.h>
#include <cstdint>

// ---------------------------------------------------------------------------
// Problem constants (from reference)
// ---------------------------------------------------------------------------
#define BB   16
#define TT   1024
#define NN   16
#define LL   100
#define DD   1024
#define HH   1024
#define PP   120            // C(16,2)
#define MROW (BB * PP)      // 1920 MLP rows
#define K1   (3 * DD)       // 3072
#define NEGV (-9000000000.0f)

typedef __attribute__((ext_vector_type(16))) __bf16 v16bf;
typedef __attribute__((ext_vector_type(8)))  __bf16 v8bf;
typedef __attribute__((ext_vector_type(4)))  __bf16 v4bf;
typedef __attribute__((ext_vector_type(8)))  float  v8f;

// ---------------------------------------------------------------------------
// WMMA helpers (CDNA5: v_wmma_f32_16x16x32_bf16, wave32)
// A fragment: 16x32 row-major source (stride elems). lane ln holds row ln&15,
//   K = kbase + (ln<16?0:8) + {0..7, 16..23}  -> two 16B loads.
// B fragment: 32x16, column n contiguous over K (we store B transposed, NxK).
//   lane ln holds col ln&15, K = kbase + (ln<16?0:8) + {0..7, 16..23} -> two 16B loads.
// ---------------------------------------------------------------------------
__device__ __forceinline__ v16bf load_frag_a(const __bf16* base, int stride, int ln) {
    int m    = ln & 15;
    int koff = (ln < 16) ? 0 : 8;
    const __bf16* p = base + (size_t)m * stride + koff;
    v8bf lo = *(const v8bf*)p;
    v8bf hi = *(const v8bf*)(p + 16);
    v16bf r;
#pragma unroll
    for (int e = 0; e < 8; ++e) { r[e] = lo[e]; r[8 + e] = hi[e]; }
    return r;
}

__device__ __forceinline__ v16bf load_frag_b(const __bf16* baseT, int stride, int ln) {
    int n  = ln & 15;
    int kb = (ln < 16) ? 0 : 8;
    const __bf16* p = baseT + (size_t)n * stride + kb;
    v8bf lo = *(const v8bf*)p;
    v8bf hi = *(const v8bf*)(p + 16);
    v16bf r;
#pragma unroll
    for (int e = 0; e < 8; ++e) { r[e] = lo[e]; r[8 + e] = hi[e]; }
    return r;
}

__device__ __forceinline__ v8f wmma_bf16(v16bf a, v16bf b, v8f c) {
    return __builtin_amdgcn_wmma_f32_16x16x32_bf16(
        /*neg_a=*/false, a, /*neg_b=*/false, b,
        /*c_mod=*/(short)0, c, /*reuse_a=*/false, /*reuse_b=*/false);
}

// ---------------------------------------------------------------------------
// fp32 -> bf16 conversion (vectorized 4-wide)
// ---------------------------------------------------------------------------
__global__ void cvt_f32_bf16_kernel(const float* __restrict__ src,
                                    __bf16* __restrict__ dst, int n4) {
    int i = blockIdx.x * blockDim.x + threadIdx.x;
    if (i >= n4) return;
    float4 v = ((const float4*)src)[i];
    v4bf o;
    o[0] = (__bf16)v.x; o[1] = (__bf16)v.y; o[2] = (__bf16)v.z; o[3] = (__bf16)v.w;
    ((v4bf*)dst)[i] = o;
}

// Transpose fp32 (K x N) -> bf16 (N x K)
__global__ void transpose_to_bf16_kernel(const float* __restrict__ W,
                                         __bf16* __restrict__ Wt, int K, int N) {
    int idx = blockIdx.x * blockDim.x + threadIdx.x;
    if (idx >= K * N) return;
    int k = idx / N, n = idx % N;
    Wt[(size_t)n * K + k] = (__bf16)W[idx];
}

// ---------------------------------------------------------------------------
// Mask lengths: masks are prefix masks; len = popcount of row.
// lens[0..15]=tlen, [16..31]=nlen, [32..47]=llen
// ---------------------------------------------------------------------------
__global__ __launch_bounds__(256) void lens_kernel(const int* __restrict__ tm,
                                                   const int* __restrict__ nm,
                                                   const int* __restrict__ im,
                                                   int* __restrict__ lens) {
    int ln = threadIdx.x & 31, wv = threadIdx.x >> 5;
    int w = blockIdx.x * 8 + wv;
    if (w >= 48) return;
    const int* src; int len;
    if (w < 16)      { src = tm + (size_t)w * TT;        len = TT; }
    else if (w < 32) { src = nm + (size_t)(w - 16) * NN; len = NN; }
    else             { src = im + (size_t)(w - 32) * LL; len = LL; }
    int s = 0;
    for (int k = ln; k < len; k += 32) s += src[k];
#pragma unroll
    for (int m = 16; m >= 1; m >>= 1) s += __shfl_xor(s, m, 32);
    if (ln == 0) lens[w] = s;
}

// ---------------------------------------------------------------------------
// grounding_scores (B,N,L): att diag with mask->NEG.
// One wave per (b, l-tile). 16x16 C tile, K loop over D.
// ---------------------------------------------------------------------------
__global__ __launch_bounds__(32) void grounding_kernel(const __bf16* __restrict__ spanb,
                                                       const __bf16* __restrict__ imageb,
                                                       const int* __restrict__ lens,
                                                       float* __restrict__ out) {
    int b = blockIdx.x / 7, lt = blockIdx.x % 7;
    int ln = threadIdx.x & 31;
    int nlen = lens[16 + b], llen = lens[32 + b];
    const __bf16* A  = spanb  + (size_t)b * NN * DD;
    const __bf16* Bt = imageb + (size_t)b * LL * DD + (size_t)(lt * 16) * DD;
    v8f acc = {};
    for (int k0 = 0; k0 < DD; k0 += 32) {
        v16bf a  = load_frag_a(A + k0, DD, ln);
        v16bf bb = load_frag_b(Bt + k0, DD, ln);
        acc = wmma_bf16(a, bb, acc);
    }
    int l = lt * 16 + (ln & 15);
    if (l < LL) {
#pragma unroll
        for (int v = 0; v < 8; ++v) {
            int n = v + ((ln >= 16) ? 8 : 0);
            float d = acc[v];
            bool valid = (n < nlen) && (l < llen);
            out[1 + (size_t)b * NN * LL + (size_t)n * LL + l] =
                (valid && d != 0.0f) ? d : NEGV;
        }
    }
}

// ---------------------------------------------------------------------------
// S[i,j] sent scores. One block (8 waves) per (i,j). Each wave computes
// 16x112 att tiles via WMMA; row-softmax (over L) is local to a tile,
// column-softmax (over T) uses online (flash) accumulators per lane.
// ---------------------------------------------------------------------------
__global__ __launch_bounds__(256) void sent_scores_kernel(const __bf16* __restrict__ textb,
                                                          const __bf16* __restrict__ imageb,
                                                          const int* __restrict__ lens,
                                                          float* __restrict__ Sout) {
    __shared__ float sm[8][112], ss[8][112], st[8][112];
    __shared__ float sp1[8];
    const int i  = blockIdx.x >> 4, j = blockIdx.x & 15;
    const int ln = threadIdx.x & 31, wv = threadIdx.x >> 5;
    const int tlen = lens[i], llen = lens[32 + j];
    const __bf16* At = textb  + (size_t)i * TT * DD;
    const __bf16* Bt = imageb + (size_t)j * LL * DD;

    float cm[7], cs[7], ct[7];
#pragma unroll
    for (int lt = 0; lt < 7; ++lt) { cm[lt] = -3.0e38f; cs[lt] = 0.f; ct[lt] = 0.f; }
    float p1 = 0.f;

    for (int t = wv; t < TT / 16; t += 8) {
        const int row0 = t * 16;
        v8f acc[7];
#pragma unroll
        for (int lt = 0; lt < 7; ++lt) acc[lt] = (v8f){0.f,0.f,0.f,0.f,0.f,0.f,0.f,0.f};

        for (int k0 = 0; k0 < DD; k0 += 32) {
            v16bf a = load_frag_a(At + (size_t)row0 * DD + k0, DD, ln);
            __builtin_prefetch(At + (size_t)row0 * DD + k0 + 64, 0, 1);
#pragma unroll
            for (int lt = 0; lt < 7; ++lt) {
                v16bf b = load_frag_b(Bt + (size_t)(lt * 16) * DD + k0, DD, ln);
                acc[lt] = wmma_bf16(a, b, acc[lt]);
            }
        }

#pragma unroll
        for (int v = 0; v < 8; ++v) {
            int  rg = row0 + v + ((ln >= 16) ? 8 : 0);
            bool rowvalid = rg < tlen;
            float xv[7];
            float rmax = -3.0e38f;
#pragma unroll
            for (int lt = 0; lt < 7; ++lt) {
                int l = lt * 16 + (ln & 15);
                float d = acc[lt][v];
                float x = (rowvalid && (l < llen) && (d != 0.0f)) ? d : NEGV;
                xv[lt] = x;
                rmax = fmaxf(rmax, x);
            }
#pragma unroll
            for (int s = 1; s < 16; s <<= 1) rmax = fmaxf(rmax, __shfl_xor(rmax, s, 32));
            float se = 0.f, sea = 0.f;
#pragma unroll
            for (int lt = 0; lt < 7; ++lt) {
                float e = __expf(xv[lt] - rmax);
                se += e; sea += e * xv[lt];
            }
#pragma unroll
            for (int s = 1; s < 16; s <<= 1) { se += __shfl_xor(se, s, 32); sea += __shfl_xor(sea, s, 32); }
            if (rowvalid && ((ln & 15) == 0)) p1 += sea / se;

            // online column-softmax accumulators (flash style)
#pragma unroll
            for (int lt = 0; lt < 7; ++lt) {
                float x  = xv[lt];
                float nm = fmaxf(cm[lt], x);
                float sc = __expf(cm[lt] - nm);
                float ex = __expf(x - nm);
                cs[lt] = cs[lt] * sc + ex;
                ct[lt] = ct[lt] * sc + ex * x;
                cm[lt] = nm;
            }
        }
    }

    // merge the two half-wave row subsets (lane l <-> l+16 hold same column)
#pragma unroll
    for (int lt = 0; lt < 7; ++lt) {
        float om = __shfl_xor(cm[lt], 16, 32);
        float os = __shfl_xor(cs[lt], 16, 32);
        float ot = __shfl_xor(ct[lt], 16, 32);
        float nm = fmaxf(cm[lt], om);
        float s0 = __expf(cm[lt] - nm), s1 = __expf(om - nm);
        cs[lt] = cs[lt] * s0 + os * s1;
        ct[lt] = ct[lt] * s0 + ot * s1;
        cm[lt] = nm;
    }
    if (ln < 16) {
#pragma unroll
        for (int lt = 0; lt < 7; ++lt) {
            sm[wv][lt * 16 + ln] = cm[lt];
            ss[wv][lt * 16 + ln] = cs[lt];
            st[wv][lt * 16 + ln] = ct[lt];
        }
    }
    p1 += __shfl_xor(p1, 16, 32);
    if (ln == 0) sp1[wv] = p1;
    __syncthreads();

    if (wv == 0) {
        float part2 = 0.f;
        for (int l = ln; l < 112; l += 32) {
            float m = -3.0e38f, s = 0.f, tt = 0.f;
#pragma unroll
            for (int w = 0; w < 8; ++w) {
                float om = sm[w][l], os = ss[w][l], ot = st[w][l];
                float nm = fmaxf(m, om);
                float s0 = __expf(m - nm), s1 = __expf(om - nm);
                s = s * s0 + os * s1;
                tt = tt * s0 + ot * s1;
                m = nm;
            }
            if (l < llen) part2 += tt / s;
        }
#pragma unroll
        for (int sft = 1; sft < 32; sft <<= 1) part2 += __shfl_xor(part2, sft, 32);
        if (ln == 0) {
            float p1t = 0.f;
#pragma unroll
            for (int w = 0; w < 8; ++w) p1t += sp1[w];
            Sout[blockIdx.x] = p1t / (float)tlen + part2 / (float)llen;
        }
    }
}

// ---------------------------------------------------------------------------
// loss from S (16x16): symmetric InfoNCE-style diag log-softmax
// ---------------------------------------------------------------------------
__global__ __launch_bounds__(32) void loss_kernel(const float* __restrict__ S,
                                                  float* __restrict__ out) {
    int ln = threadIdx.x;
    float contrib = 0.f;
    if (ln < 16) {
        int i = ln;
        float m = -3.0e38f;
        for (int jj = 0; jj < 16; ++jj) m = fmaxf(m, S[i * 16 + jj]);
        float se = 0.f;
        for (int jj = 0; jj < 16; ++jj) se += __expf(S[i * 16 + jj] - m);
        float diag_r = S[i * 17] - (m + __logf(se));
        m = -3.0e38f;
        for (int jj = 0; jj < 16; ++jj) m = fmaxf(m, S[jj * 16 + i]);
        se = 0.f;
        for (int jj = 0; jj < 16; ++jj) se += __expf(S[jj * 16 + i] - m);
        float diag_c = S[i * 17] - (m + __logf(se));
        contrib = -(diag_r + diag_c);
    }
#pragma unroll
    for (int s = 1; s < 32; s <<= 1) contrib += __shfl_xor(contrib, s, 32);
    if (ln == 0) out[0] = contrib / (float)BB;
}

// ---------------------------------------------------------------------------
// Build x = [first, second, first*second] (1920 x 3072) in bf16.
// Row r = p*16 + n ; first = span[fi[p], n, :], second = span[si[p], n, :]
// (reference indexes the BATCH axis with pair indices -- replicated).
// ---------------------------------------------------------------------------
__device__ __forceinline__ void pair_from_p(int p, int& a, int& b) {
    int i = 0, rem = p, cnt = NN - 1;
    while (rem >= cnt) { rem -= cnt; ++i; --cnt; }
    a = i; b = i + 1 + rem;
}

__global__ void build_x_kernel(const float* __restrict__ span, __bf16* __restrict__ x) {
    int idx = blockIdx.x * blockDim.x + threadIdx.x;
    const int total = MROW * K1;
    if (idx >= total) return;
    int r = idx / K1, k = idx % K1;
    int p = r >> 4, n = r & 15;
    int a, b;
    pair_from_p(p, a, b);
    float val;
    if (k < DD)            val = span[((size_t)a * NN + n) * DD + k];
    else if (k < 2 * DD)   val = span[((size_t)b * NN + n) * DD + (k - DD)];
    else {
        int kk = k - 2 * DD;
        val = span[((size_t)a * NN + n) * DD + kk] * span[((size_t)b * NN + n) * DD + kk];
    }
    x[idx] = (__bf16)val;
}

// ---------------------------------------------------------------------------
// Generic bf16 WMMA GEMM: C = act(A(MxK) @ Bt^T + bias). Bt is (N x K).
// One wave per 16x16 output tile; 8 waves per block.
// ---------------------------------------------------------------------------
__global__ __launch_bounds__(256) void gemm_bf16_kernel(const __bf16* __restrict__ A,
                                                        const __bf16* __restrict__ Bt,
                                                        const float* __restrict__ bias,
                                                        float* __restrict__ Cf,
                                                        __bf16* __restrict__ Cb,
                                                        int M, int N, int K, int relu) {
    int ln = threadIdx.x & 31, wv = threadIdx.x >> 5;
    int tile = blockIdx.x * 8 + wv;
    int ntiles = N >> 4;
    int mt = tile / ntiles, nt = tile % ntiles;
    if (mt * 16 >= M) return;
    const __bf16* Ab = A + (size_t)mt * 16 * K;
    const __bf16* Bb = Bt + (size_t)nt * 16 * K;
    v8f acc = {};
    for (int k0 = 0; k0 < K; k0 += 32) {
        v16bf a = load_frag_a(Ab + k0, K, ln);
        v16bf b = load_frag_b(Bb + k0, K, ln);
        __builtin_prefetch(Bb + k0 + 64, 0, 1);
        acc = wmma_bf16(a, b, acc);
    }
    int col = nt * 16 + (ln & 15);
    float bv = bias ? bias[col] : 0.f;
#pragma unroll
    for (int v = 0; v < 8; ++v) {
        int row = mt * 16 + v + ((ln >= 16) ? 8 : 0);
        float xx = acc[v] + bv;
        if (relu) xx = fmaxf(xx, 0.f);
        size_t off = (size_t)row * N + col;
        if (Cf) Cf[off] = xx;
        if (Cb) Cb[off] = (__bf16)xx;
    }
}

// Final layer: 1920 dot products of length 1024 against W3, + b3.
__global__ __launch_bounds__(256) void final_dot_kernel(const float* __restrict__ h2,
                                                        const float* __restrict__ W3,
                                                        const float* __restrict__ b3,
                                                        float* __restrict__ out) {
    int ln = threadIdx.x & 31, wv = threadIdx.x >> 5;
    int r = blockIdx.x * 8 + wv;
    if (r >= MROW) return;
    float s = 0.f;
    for (int k = ln; k < HH; k += 32) s += h2[(size_t)r * HH + k] * W3[k];
#pragma unroll
    for (int m = 16; m >= 1; m >>= 1) s += __shfl_xor(s, m, 32);
    if (ln == 0) out[r] = s + b3[0];
}

// ---------------------------------------------------------------------------
// Host launcher
// ---------------------------------------------------------------------------
extern "C" void kernel_launch(void* const* d_in, const int* in_sizes, int n_in,
                              void* d_out, int out_size, void* d_ws, size_t ws_size,
                              hipStream_t stream) {
    const float* text  = (const float*)d_in[0];
    const float* span  = (const float*)d_in[1];
    const float* image = (const float*)d_in[2];
    const int*   tmask = (const int*)d_in[3];
    const int*   nmask = (const int*)d_in[4];
    const int*   imask = (const int*)d_in[5];
    const float* W1 = (const float*)d_in[6];
    const float* b1 = (const float*)d_in[7];
    const float* W2 = (const float*)d_in[8];
    const float* b2 = (const float*)d_in[9];
    const float* W3 = (const float*)d_in[10];
    const float* b3 = (const float*)d_in[11];
    float* out = (float*)d_out;

    char* ws = (char*)d_ws;
    size_t off = 0;
    auto alloc = [&](size_t bytes) -> char* {
        char* p = ws + off;
        off += (bytes + 255) & ~(size_t)255;
        return p;
    };
    __bf16* textb  = (__bf16*)alloc((size_t)BB * TT * DD * 2);   // 32 MB
    __bf16* imageb = (__bf16*)alloc((size_t)BB * LL * DD * 2);   // 3.2 MB
    __bf16* spanb  = (__bf16*)alloc((size_t)BB * NN * DD * 2);   // 0.5 MB
    __bf16* Wt1    = (__bf16*)alloc((size_t)K1 * HH * 2);        // 6 MB (HH x K1)
    __bf16* Wt2    = (__bf16*)alloc((size_t)HH * HH * 2);        // 2 MB
    __bf16* xb     = (__bf16*)alloc((size_t)MROW * K1 * 2);      // 11.8 MB
    __bf16* h1b    = (__bf16*)alloc((size_t)MROW * HH * 2);      // 3.9 MB
    float*  h2f    = (float*)alloc((size_t)MROW * HH * 4);       // 7.9 MB
    float*  Smat   = (float*)alloc(256 * 4);
    int*    lens   = (int*)alloc(48 * 4);
    (void)ws_size; (void)in_sizes; (void)n_in; (void)out_size;

    // 1) precision conversion + weight transposes
    {
        int n4 = (BB * TT * DD) / 4;
        cvt_f32_bf16_kernel<<<(n4 + 255) / 256, 256, 0, stream>>>(text, textb, n4);
    }
    {
        int n4 = (BB * LL * DD) / 4;
        cvt_f32_bf16_kernel<<<(n4 + 255) / 256, 256, 0, stream>>>(image, imageb, n4);
    }
    {
        int n4 = (BB * NN * DD) / 4;
        cvt_f32_bf16_kernel<<<(n4 + 255) / 256, 256, 0, stream>>>(span, spanb, n4);
    }
    transpose_to_bf16_kernel<<<(K1 * HH + 255) / 256, 256, 0, stream>>>(W1, Wt1, K1, HH);
    transpose_to_bf16_kernel<<<(HH * HH + 255) / 256, 256, 0, stream>>>(W2, Wt2, HH, HH);

    // 2) mask lengths
    lens_kernel<<<6, 256, 0, stream>>>(tmask, nmask, imask, lens);

    // 3) grounding scores (output slot 2)
    grounding_kernel<<<BB * 7, 32, 0, stream>>>(spanb, imageb, lens, out);

    // 4) sentence alignment matrix S, then loss (output slot 1)
    sent_scores_kernel<<<BB * BB, 256, 0, stream>>>(textb, imageb, lens, Smat);
    loss_kernel<<<1, 32, 0, stream>>>(Smat, out);

    // 5) pairwise MLP (output slot 3)
    build_x_kernel<<<(MROW * K1 + 255) / 256, 256, 0, stream>>>(span, xb);
    gemm_bf16_kernel<<<(MROW / 16) * (HH / 16) / 8, 256, 0, stream>>>(
        xb, Wt1, b1, nullptr, h1b, MROW, HH, K1, 1);
    gemm_bf16_kernel<<<(MROW / 16) * (HH / 16) / 8, 256, 0, stream>>>(
        h1b, Wt2, b2, h2f, nullptr, MROW, HH, HH, 1);
    final_dot_kernel<<<MROW / 8, 256, 0, stream>>>(h2f, W3, b3, out + 1 + BB * NN * LL);
}
